// GloballyAttentiveDecoder_61168924229948
// MI455X (gfx1250) — matrix operand order repaired and checked by reading, not compiled, not run on hardware
//
#include <hip/hip_runtime.h>
#include <math.h>

// Problem dims (compile-time, from the reference)
#define VDIM   32000
#define HDIM   1024
#define IMGDIM 2048
#define BDIM   256
#define SDIM   64

typedef __attribute__((ext_vector_type(16))) __bf16 v16bf;
typedef __attribute__((ext_vector_type(2)))  __bf16 v2bf;
typedef __attribute__((ext_vector_type(8)))  float  v8f;

union V16U { v16bf v; unsigned short u[16]; };

// round-to-nearest-even f32 pair -> packed bf16x2 in one dword
__device__ __forceinline__ unsigned int pack2bf(float lo, float hi) {
#if __has_builtin(__builtin_amdgcn_cvt_pk_bf16_f32)
  union { v2bf v; unsigned int u; } cv;
  cv.v = __builtin_amdgcn_cvt_pk_bf16_f32(lo, hi);   // v_cvt_pk_bf16_f32
  return cv.u;
#else
  unsigned int a = __float_as_uint(lo);
  unsigned int b = __float_as_uint(hi);
  a = (a + 0x7FFFu + ((a >> 16) & 1u)) >> 16;          // low 16
  b = (b + 0x7FFFu + ((b >> 16) & 1u)) & 0xFFFF0000u;  // high 16
  return a | b;
#endif
}

// Epilogue modes
#define EPI_NONE 0
#define EPI_BIAS 1
#define EPI_EMB  2   // x = 0.5*(emb_table[ids[m]] + (A@B + bias))

// C[M,N] (+)= A[M,K] x B, where B(n,k) = Bw[bOff + n*bStrideN + k*bStrideK].
// BK_CONTIG: bStrideK == 1 (vectorized B loads, no stride math in the loop).
// Block: 256 threads (8 wave32), tile 64(M) x 128(N), waves 2x4, wave tile 32x32
// (2x2 WMMA 16x16 subtiles), K-step 32, bf16 operands staged through LDS.
// Software pipelined: the global loads for tile k+1 issue right after the
// tile-ready barrier, overlapping HBM latency with ds_load + v_wmma work.
template <bool BK_CONTIG>
__global__ __launch_bounds__(256)
void wmma_gemm_bf16(const float* __restrict__ A, int lda,
                    const float* __restrict__ Bw, long bStrideN, long bStrideK, long bOff,
                    float* __restrict__ C, int ldc,
                    const float* __restrict__ bias,
                    const int* __restrict__ ids,
                    const float* __restrict__ emb,
                    int M, int N, int K, int mode, int accumulate)
{
  __shared__ __align__(16) unsigned short ldsA[64][40];   // 64x32 bf16 + pad
  __shared__ __align__(16) unsigned short ldsB[128][40];  // 128x32 bf16 + pad

  const int tid   = threadIdx.x;
  const int lane  = tid & 31;
  const int wave  = tid >> 5;
  const int waveM = (wave & 1) * 32;
  const int waveN = (wave >> 1) * 32;
  const int tileM = blockIdx.y * 64;
  const int tileN = blockIdx.x * 128;
  const int g     = lane >> 4;    // lane group (ISA A/B operand layout)
  const int l15   = lane & 15;

  v8f acc[2][2];
  #pragma unroll
  for (int i = 0; i < 2; ++i)
    #pragma unroll
    for (int j = 0; j < 2; ++j)
      #pragma unroll
      for (int r = 0; r < 8; ++r)
        acc[i][j][r] = 0.0f;

  const int aRow = tid >> 2;        // 0..63
  const int aCol = (tid & 3) * 8;   // 0,8,16,24
  const int bRow = tid >> 1;        // 0..127
  const int bCol = (tid & 1) * 16;  // 0,16

  // Hoisted global pointers; stepped each K iteration (no per-iter mul).
  const float* aPtr = A + (size_t)(tileM + aRow) * lda + aCol;
  const float* bPtr = Bw + bOff + (size_t)(tileN + bRow) * bStrideN
                               + (size_t)bCol * bStrideK;
  const long bStep = 32L * bStrideK;

  // Invariant LDS operand addresses (two 16B runs per operand).
  const unsigned short* aLds0 = &ldsA[waveM + l15][g * 8];
  const unsigned short* aLds1 = &ldsA[waveM + 16 + l15][g * 8];
  const unsigned short* bLds0 = &ldsB[waveN + l15][g * 8];
  const unsigned short* bLds1 = &ldsB[waveN + 16 + l15][g * 8];

  // Register-staged tile fragments (prefetch destination).
  float4 fa0, fa1;
  float4 fb0, fb1, fb2, fb3;   // BK_CONTIG path
  float  sb[16];               // strided-K path

  auto loadFrags = [&]() {
    fa0 = ((const float4*)aPtr)[0];
    fa1 = ((const float4*)aPtr)[1];
    if constexpr (BK_CONTIG) {
      fb0 = ((const float4*)bPtr)[0];
      fb1 = ((const float4*)bPtr)[1];
      fb2 = ((const float4*)bPtr)[2];
      fb3 = ((const float4*)bPtr)[3];
    } else {
      #pragma unroll
      for (int h = 0; h < 16; ++h)
        sb[h] = bPtr[(size_t)h * bStrideK];
    }
  };

  auto storeFrags = [&]() {
    uint4 pa;
    pa.x = pack2bf(fa0.x, fa0.y);
    pa.y = pack2bf(fa0.z, fa0.w);
    pa.z = pack2bf(fa1.x, fa1.y);
    pa.w = pack2bf(fa1.z, fa1.w);
    *(uint4*)&ldsA[aRow][aCol] = pa;
    if constexpr (BK_CONTIG) {
      uint4 pb0, pb1;
      pb0.x = pack2bf(fb0.x, fb0.y);
      pb0.y = pack2bf(fb0.z, fb0.w);
      pb0.z = pack2bf(fb1.x, fb1.y);
      pb0.w = pack2bf(fb1.z, fb1.w);
      pb1.x = pack2bf(fb2.x, fb2.y);
      pb1.y = pack2bf(fb2.z, fb2.w);
      pb1.z = pack2bf(fb3.x, fb3.y);
      pb1.w = pack2bf(fb3.z, fb3.w);
      *(uint4*)&ldsB[bRow][bCol]     = pb0;
      *(uint4*)&ldsB[bRow][bCol + 8] = pb1;
    } else {
      #pragma unroll
      for (int h = 0; h < 8; ++h)
        ((unsigned int*)&ldsB[bRow][bCol])[h] = pack2bf(sb[2 * h], sb[2 * h + 1]);
    }
  };

  loadFrags();  // tile 0

  for (int k0 = 0; k0 < K; k0 += 32) {
    __syncthreads();   // prior consumers done
    storeFrags();
    __syncthreads();   // tile ready

    // Prefetch tile k0+32 into registers; loadcnt is only waited at the
    // next iteration's storeFrags, so HBM latency overlaps the math below.
    if (k0 + 32 < K) {
      aPtr += 32;
      bPtr += bStep;
      loadFrags();
    }

    // ---- build WMMA operands per ISA lane layout ----
    // A 16x32 bf16: lane l15 = row M; group g: VGPR0-3 K=g*8+0..7,
    // VGPR4-7 K=16+g*8+0..7  -> two contiguous 8-half runs (ds_load_b128).
    V16U aop[2], bop[2];
    *(uint4*)&aop[0].u[0] = *(const uint4*)(aLds0);
    *(uint4*)&aop[0].u[8] = *(const uint4*)(aLds0 + 16);
    *(uint4*)&aop[1].u[0] = *(const uint4*)(aLds1);
    *(uint4*)&aop[1].u[8] = *(const uint4*)(aLds1 + 16);
    // B 32x16 bf16: lane l15 = col N; group g holds K = g*8+0..7, 16+g*8+0..7.
    *(uint4*)&bop[0].u[0] = *(const uint4*)(bLds0);
    *(uint4*)&bop[0].u[8] = *(const uint4*)(bLds0 + 16);
    *(uint4*)&bop[1].u[0] = *(const uint4*)(bLds1);
    *(uint4*)&bop[1].u[8] = *(const uint4*)(bLds1 + 16);

    #pragma unroll
    for (int i = 0; i < 2; ++i)
      #pragma unroll
      for (int j = 0; j < 2; ++j)
        acc[i][j] = __builtin_amdgcn_wmma_f32_16x16x32_bf16(
            false, aop[i].v, false, bop[j].v, (short)0, acc[i][j], false, false);
  }

  // ---- epilogue + store: C/D layout: VGPR r -> M = r + 8*g, lane l15 -> N ----
  #pragma unroll
  for (int i = 0; i < 2; ++i) {
    #pragma unroll
    for (int j = 0; j < 2; ++j) {
      const int col = tileN + waveN + j * 16 + l15;
      #pragma unroll
      for (int r = 0; r < 8; ++r) {
        const int row = tileM + waveM + i * 16 + g * 8 + r;
        float v = acc[i][j][r];
        const size_t idx = (size_t)row * ldc + col;
        if (accumulate) v += C[idx];
        if (mode >= EPI_BIAS && bias) v += bias[col];
        if (mode == EPI_EMB) v = 0.5f * (v + emb[(size_t)ids[row] * HDIM + col]);
        C[idx] = v;
      }
    }
  }
}

// GRU gate nonlinearity: h_new = (1-z)*n + z*h  (gates r,z,n PyTorch layout)
__global__ __launch_bounds__(256)
void gru_gate_kernel(const float* __restrict__ gi, const float* __restrict__ gh,
                     const float* __restrict__ hprev, float* __restrict__ hnew)
{
  const int b = blockIdx.x;
  const size_t gbase = (size_t)b * 3 * HDIM;
  for (int h = threadIdx.x; h < HDIM; h += 256) {
    float ir = gi[gbase + h], iz = gi[gbase + HDIM + h], in = gi[gbase + 2 * HDIM + h];
    float hr = gh[gbase + h], hz = gh[gbase + HDIM + h], hn = gh[gbase + 2 * HDIM + h];
    float r = 1.0f / (1.0f + expf(-(ir + hr)));
    float z = 1.0f / (1.0f + expf(-(iz + hz)));
    float n = tanhf(in + r * hn);
    float hp = hprev[(size_t)b * HDIM + h];
    hnew[(size_t)b * HDIM + h] = (1.0f - z) * n + z * hp;
  }
}

// score[b,s] = tmpW[b,:] . enc[b,s,:] ; softmax over s ; ctx[b,:] = attn . enc[b]
__global__ __launch_bounds__(256)
void attn_kernel(const float* __restrict__ tmpW, const float* __restrict__ enc,
                 float* __restrict__ attn_out, float* __restrict__ ctx)
{
  __shared__ float sTmp[HDIM];
  __shared__ float sScore[SDIM];
  const int b = blockIdx.x;
  const int tid = threadIdx.x, lane = tid & 31, wave = tid >> 5;

  for (int i = tid; i < HDIM; i += 256) sTmp[i] = tmpW[(size_t)b * HDIM + i];
  __syncthreads();

  for (int s = wave; s < SDIM; s += 8) {
    const float* e = enc + ((size_t)b * SDIM + s) * HDIM;
    float a = 0.0f;
    for (int k = lane; k < HDIM; k += 32) a += sTmp[k] * e[k];
    #pragma unroll
    for (int o = 16; o > 0; o >>= 1) a += __shfl_xor(a, o, 32);
    if (lane == 0) sScore[s] = a;
  }
  __syncthreads();

  if (tid == 0) {  // S=64: trivial serial softmax
    float mx = sScore[0];
    for (int s = 1; s < SDIM; ++s) mx = fmaxf(mx, sScore[s]);
    float sum = 0.0f;
    for (int s = 0; s < SDIM; ++s) { float e2 = expf(sScore[s] - mx); sScore[s] = e2; sum += e2; }
    float inv = 1.0f / sum;
    for (int s = 0; s < SDIM; ++s) sScore[s] *= inv;
  }
  __syncthreads();

  if (tid < SDIM) attn_out[(size_t)b * SDIM + tid] = sScore[tid];

  for (int h = tid; h < HDIM; h += 256) {
    float a = 0.0f;
    #pragma unroll 4
    for (int s = 0; s < SDIM; ++s)
      a += sScore[s] * enc[((size_t)b * SDIM + s) * HDIM + h];
    ctx[(size_t)b * HDIM + h] = a;
  }
}

// in-place log_softmax over V per row
__global__ __launch_bounds__(256)
void log_softmax_kernel(float* __restrict__ logits)
{
  __shared__ float redMax[8];
  __shared__ float redSum[8];
  const int b = blockIdx.x;
  float* row = logits + (size_t)b * VDIM;
  const int tid = threadIdx.x, lane = tid & 31, wave = tid >> 5;

  float m = -3.402823466e+38f;
  for (int i = tid; i < VDIM; i += 256) m = fmaxf(m, row[i]);
  #pragma unroll
  for (int o = 16; o > 0; o >>= 1) m = fmaxf(m, __shfl_xor(m, o, 32));
  if (lane == 0) redMax[wave] = m;
  __syncthreads();
  float mx = redMax[0];
  #pragma unroll
  for (int w = 1; w < 8; ++w) mx = fmaxf(mx, redMax[w]);

  float s = 0.0f;
  for (int i = tid; i < VDIM; i += 256) s += expf(row[i] - mx);
  #pragma unroll
  for (int o = 16; o > 0; o >>= 1) s += __shfl_xor(s, o, 32);
  if (lane == 0) redSum[wave] = s;
  __syncthreads();
  float total = 0.0f;
  #pragma unroll
  for (int w = 0; w < 8; ++w) total += redSum[w];
  const float ls = mx + logf(total);

  for (int i = tid; i < VDIM; i += 256) row[i] -= ls;
}

extern "C" void kernel_launch(void* const* d_in, const int* in_sizes, int n_in,
                              void* d_out, int out_size, void* d_ws, size_t ws_size,
                              hipStream_t stream)
{
  const int*   ids   = (const int*)  d_in[0];
  const float* img   = (const float*)d_in[1];
  const float* hid   = (const float*)d_in[2];   // [1,B,H] == [B,H]
  const float* enc   = (const float*)d_in[3];
  const float* embT  = (const float*)d_in[4];
  const float* lin_w = (const float*)d_in[5];
  const float* lin_b = (const float*)d_in[6];
  const float* attnW = (const float*)d_in[7];
  const float* W_ih  = (const float*)d_in[8];
  const float* W_hh  = (const float*)d_in[9];
  const float* b_ih  = (const float*)d_in[10];
  const float* b_hh  = (const float*)d_in[11];
  const float* combW = (const float*)d_in[12];
  const float* combB = (const float*)d_in[13];
  const float* outW  = (const float*)d_in[14];
  const float* outB  = (const float*)d_in[15];

  // Outputs flat in return order: logp [B,V], h_new [1,B,H], attn [B,S]
  float* logp  = (float*)d_out;
  float* hnew  = logp + (size_t)BDIM * VDIM;
  float* attnO = hnew + (size_t)BDIM * HDIM;

  // Scratch layout (~10.5 MB)
  float* x    = (float*)d_ws;
  float* gi   = x    + (size_t)BDIM * HDIM;
  float* gh   = gi   + (size_t)BDIM * 3 * HDIM;
  float* tmpW = gh   + (size_t)BDIM * 3 * HDIM;
  float* ctx  = tmpW + (size_t)BDIM * HDIM;
  float* comb = ctx  + (size_t)BDIM * HDIM;

  dim3 blk(256);

  // 1) x = 0.5*(emb_table[ids] + img @ lin_w.T + lin_b)
  wmma_gemm_bf16<true><<<dim3(HDIM / 128, BDIM / 64), blk, 0, stream>>>(
      img, IMGDIM, lin_w, (long)IMGDIM, 1L, 0L, x, HDIM,
      lin_b, ids, embT, BDIM, HDIM, IMGDIM, EPI_EMB, 0);

  // 2) gi = x @ W_ih.T + b_ih
  wmma_gemm_bf16<true><<<dim3(3 * HDIM / 128, BDIM / 64), blk, 0, stream>>>(
      x, HDIM, W_ih, (long)HDIM, 1L, 0L, gi, 3 * HDIM,
      b_ih, nullptr, nullptr, BDIM, 3 * HDIM, HDIM, EPI_BIAS, 0);

  // 3) gh = h @ W_hh.T + b_hh
  wmma_gemm_bf16<true><<<dim3(3 * HDIM / 128, BDIM / 64), blk, 0, stream>>>(
      hid, HDIM, W_hh, (long)HDIM, 1L, 0L, gh, 3 * HDIM,
      b_hh, nullptr, nullptr, BDIM, 3 * HDIM, HDIM, EPI_BIAS, 0);

  // 4) GRU gates -> h_new (written straight into its d_out slice)
  gru_gate_kernel<<<dim3(BDIM), blk, 0, stream>>>(gi, gh, hid, hnew);

  // 5) tmpW = h_new @ attn_W   (B(n,k) = attnW[k*H + n] -> strideN=1, strideK=H)
  wmma_gemm_bf16<false><<<dim3(HDIM / 128, BDIM / 64), blk, 0, stream>>>(
      hnew, HDIM, attnW, 1L, (long)HDIM, 0L, tmpW, HDIM,
      nullptr, nullptr, nullptr, BDIM, HDIM, HDIM, EPI_NONE, 0);

  // 6) attention: score -> softmax -> ctx (enc stays hot in 192MB L2)
  attn_kernel<<<dim3(BDIM), blk, 0, stream>>>(tmpW, enc, attnO, ctx);

  // 7) comb = ctx @ comb_w[:, :H].T + comb_b
  wmma_gemm_bf16<true><<<dim3(HDIM / 128, BDIM / 64), blk, 0, stream>>>(
      ctx, HDIM, combW, (long)(2 * HDIM), 1L, 0L, comb, HDIM,
      combB, nullptr, nullptr, BDIM, HDIM, HDIM, EPI_BIAS, 0);

  // 8) comb += h_new @ comb_w[:, H:].T   (accumulate into comb)
  wmma_gemm_bf16<true><<<dim3(HDIM / 128, BDIM / 64), blk, 0, stream>>>(
      hnew, HDIM, combW, (long)(2 * HDIM), 1L, (long)HDIM, comb, HDIM,
      nullptr, nullptr, nullptr, BDIM, HDIM, HDIM, EPI_NONE, 1);

  // 9) logits = comb @ out_w.T + out_b  (the 131MB streaming GEMM)
  wmma_gemm_bf16<true><<<dim3(VDIM / 128, BDIM / 64), blk, 0, stream>>>(
      comb, HDIM, outW, (long)HDIM, 1L, 0L, logp, VDIM,
      outB, nullptr, nullptr, BDIM, VDIM, HDIM, EPI_BIAS, 0);

  // 10) logp = log_softmax(logits) in place
  log_softmax_kernel<<<dim3(BDIM), blk, 0, stream>>>(logp);
}
